// DeConv_1829656068503
// MI455X (gfx1250) — matrix operand order repaired and checked by reading, not compiled, not run on hardware
//
#include <hip/hip_runtime.h>

// ---------------------------------------------------------------- types
typedef __attribute__((ext_vector_type(16))) __bf16         v16bf;
typedef __attribute__((ext_vector_type(8)))  float          v8f;
typedef __attribute__((ext_vector_type(16))) unsigned short v16us;
typedef __attribute__((ext_vector_type(8)))  unsigned short v8us;

union BFrag { v16us u; v16bf b; };
static_assert(sizeof(v16bf) == 32, "bf16 vector size");

__device__ __forceinline__ unsigned short f2bf(float f) {
  union { float f; unsigned u; } v; v.f = f;
  unsigned r = v.u + 0x7FFFu + ((v.u >> 16) & 1u);   // round-to-nearest-even
  return (unsigned short)(r >> 16);
}

__device__ __forceinline__ v16us cat8(v8us lo, v8us hi) {
  return __builtin_shufflevector(lo, hi, 0, 1, 2, 3, 4, 5, 6, 7,
                                         8, 9, 10, 11, 12, 13, 14, 15);
}

#define WMMA_BF16(Af, Bf, Cacc)                                               \
  __builtin_amdgcn_wmma_f32_16x16x32_bf16(false, (Af), false, (Bf), (short)0, \
                                          (Cacc), false, false)

// ---------------------------------------------------------------- geometry
constexpr int Bn = 8, Cc = 256, Hh = 48, Ww = 48;
constexpr int HW = Hh * Ww;            // 2304
constexpr int N1 = Bn * HW;            // 18432 input pixels
constexpr int K1 = Cc * 9;             // 2304 (deform-conv reduction)
constexpr int Mo = 256;                // Cout
constexpr int K2 = Cc * 4;             // 1024 (per-parity transposed-conv reduction)
constexpr int HO = 96, WO = 96;
constexpr float EPS = 1e-5f;
constexpr int MB1 = Mo / 32;           // 8  m-blocks (32 wide) for GEMM1/2
constexpr int NB1 = N1 / 32;           // 576 n-blocks (32 wide)
constexpr int CHUNK = 256;             // pixels per bn1 partial block
constexpr int NCHUNK = N1 / CHUNK;     // 72

// ================================================================ 1) offset/mod conv (f32, precision-sensitive)
__global__ __launch_bounds__(256) void offmod_conv(
    const float* __restrict__ x, const float* __restrict__ w_off,
    const float* __restrict__ b_off, const float* __restrict__ w_mod,
    const float* __restrict__ b_mod, float* __restrict__ offmod) {
  int gid = blockIdx.x * blockDim.x + threadIdx.x;
  if (gid >= N1 * 27) return;
  int oc = gid % 27;
  int n  = gid / 27;
  int b = n / HW, rem = n % HW, y = rem / Ww, xw = rem % Ww;
  const float* wp = (oc < 18) ? (w_off + (size_t)oc * Cc * 9)
                              : (w_mod + (size_t)(oc - 18) * Cc * 9);
  float bias = (oc < 18) ? b_off[oc] : b_mod[oc - 18];
  float acc = 0.f;
  const float* xb = x + (size_t)b * Cc * HW;
  for (int c = 0; c < Cc; ++c) {
    const float* xc = xb + c * HW;
    const float* wc = wp + c * 9;
#pragma unroll
    for (int ky = 0; ky < 3; ++ky) {
      int yy = y + ky - 1;
      if (yy < 0 || yy >= Hh) continue;
#pragma unroll
      for (int kx = 0; kx < 3; ++kx) {
        int xx = xw + kx - 1;
        if (xx < 0 || xx >= Ww) continue;
        acc = fmaf(xc[yy * Ww + xx], wc[ky * 3 + kx], acc);
      }
    }
  }
  offmod[(size_t)n * 27 + oc] = acc + bias;
}

// ================================================================ 2) bilinear sample + modulate -> bf16 im2col V1[n][tap*256+c]
__global__ __launch_bounds__(256) void build_v1(
    const float* __restrict__ x, const float* __restrict__ offmod,
    unsigned short* __restrict__ V1) {
  int gid = blockIdx.x * blockDim.x + threadIdx.x;
  if (gid >= N1 * 9) return;
  int tap = gid % 9;
  int n   = gid / 9;
  int b = n / HW, rem = n % HW, y = rem / Ww, xw = rem % Ww;
  const float* om = offmod + (size_t)n * 27;
  float offy = om[2 * tap], offx = om[2 * tap + 1];
  float mod  = 2.f / (1.f + __expf(-om[18 + tap]));
  float py = (float)y  + (float)(tap / 3 - 1) + offy;
  float px = (float)xw + (float)(tap % 3 - 1) + offx;
  float fy = floorf(py), fx = floorf(px);
  float wy = py - fy, wx = px - fx;
  int y0 = (int)fy, x0 = (int)fx, y1 = y0 + 1, x1 = x0 + 1;
  float vy0 = (y0 >= 0 && y0 < Hh) ? 1.f : 0.f;
  float vy1 = (y1 >= 0 && y1 < Hh) ? 1.f : 0.f;
  float vx0 = (x0 >= 0 && x0 < Ww) ? 1.f : 0.f;
  float vx1 = (x1 >= 0 && x1 < Ww) ? 1.f : 0.f;
  float w00 = (1.f - wy) * (1.f - wx) * vy0 * vx0 * mod;
  float w01 = (1.f - wy) * wx         * vy0 * vx1 * mod;
  float w10 = wy * (1.f - wx)         * vy1 * vx0 * mod;
  float w11 = wy * wx                 * vy1 * vx1 * mod;
  int cy0 = min(max(y0, 0), Hh - 1), cy1 = min(max(y1, 0), Hh - 1);
  int cx0 = min(max(x0, 0), Ww - 1), cx1 = min(max(x1, 0), Ww - 1);
  int i00 = cy0 * Ww + cx0, i01 = cy0 * Ww + cx1;
  int i10 = cy1 * Ww + cx0, i11 = cy1 * Ww + cx1;
  const float* xb = x + (size_t)b * Cc * HW;
  unsigned short* vp = V1 + (size_t)n * K1 + tap * Cc;
  for (int c = 0; c < Cc; ++c) {
    const float* xc = xb + c * HW;
    float v = w00 * xc[i00] + w01 * xc[i01] + w10 * xc[i10] + w11 * xc[i11];
    vp[c] = f2bf(v);
  }
}

// ================================================================ 3) pack w_dc -> bf16 A1[o][tap*256+c]
__global__ __launch_bounds__(256) void pack_wdc(const float* __restrict__ w_dc,
                                                unsigned short* __restrict__ A1) {
  int gid = blockIdx.x * blockDim.x + threadIdx.x;
  if (gid >= Mo * K1) return;
  int o = gid / K1, k = gid % K1;
  int tap = k >> 8, c = k & 255;
  A1[gid] = f2bf(w_dc[((size_t)o * Cc + c) * 9 + tap]);
}

// ================================================================ 4) GEMM1: (256 x 2304) x (2304 x 18432), 32x32 tile per wave
// 4 independent accumulators per wave -> 4 pipelined WMMAs per k-step,
// 16 FLOP/byte of fragment traffic (2x round 1).
__global__ __launch_bounds__(256) void gemm1_wmma(
    const unsigned short* __restrict__ A,   // Mo x K1, row-major bf16
    const unsigned short* __restrict__ Bv,  // N1 x K1, row-major bf16 (== B^T)
    float* __restrict__ C) {                // N1 x Mo, row-major f32 (pixel-major)
  const int wave = blockIdx.x * (blockDim.x >> 5) + (threadIdx.x >> 5);
  const int lane = threadIdx.x & 31;
  const int mt = wave % MB1;                // 8 m-blocks of 32
  const int nt = wave / MB1;                // 576 n-blocks of 32
  const int l16 = lane & 15;
  const int hlf = lane >> 4;
  const unsigned short* a0p = A + (size_t)(mt * 32 + l16) * K1 + hlf * 16;
  const unsigned short* a1p = a0p + (size_t)16 * K1;
  const unsigned short* b0p = Bv + (size_t)(nt * 32 + l16) * K1 + hlf * 16;
  const unsigned short* b1p = b0p + (size_t)16 * K1;
  v8f acc00 = {}, acc10 = {}, acc01 = {}, acc11 = {};
  for (int kb = 0; kb < K1; kb += 32) {
    BFrag a0, a1, b0, b1;
    a0.u = *(const v16us*)(a0p + kb);
    a1.u = *(const v16us*)(a1p + kb);
    b0.u = *(const v16us*)(b0p + kb);
    b1.u = *(const v16us*)(b1p + kb);
    __builtin_prefetch(b0p + kb + 256, 0, 1);
    __builtin_prefetch(b1p + kb + 256, 0, 1);
    acc00 = WMMA_BF16(a0.b, b0.b, acc00);   // 4 independent WMMAs: XDL can
    acc10 = WMMA_BF16(a1.b, b0.b, acc10);   // pipeline them back-to-back
    acc01 = WMMA_BF16(a0.b, b1.b, acc01);
    acc11 = WMMA_BF16(a1.b, b1.b, acc11);
  }
  // C fragment: VGPR r -> M = mbase + hlf*8 + r, N = nbase + l16 (contiguous v8f)
  float* c00 = C + (size_t)(nt * 32 + l16) * Mo + mt * 32 + hlf * 8;
  *(v8f*)(c00)      = acc00;
  *(v8f*)(c00 + 16) = acc10;
  float* c01 = c00 + (size_t)16 * Mo;
  *(v8f*)(c01)      = acc01;
  *(v8f*)(c01 + 16) = acc11;
}

// ================================================================ 5a) BN1 partial sums, fully coalesced (thread t owns channel t)
__global__ __launch_bounds__(256) void bn1_partial(const float* __restrict__ out1,
                                                   float* __restrict__ part) {
  int blk = blockIdx.x;            // NCHUNK blocks
  int t = threadIdx.x;             // channel
  float s = 0.f, q = 0.f;
  const float* p = out1 + (size_t)blk * CHUNK * Mo + t;
  for (int n = 0; n < CHUNK; ++n) {
    float v = p[(size_t)n * Mo];   // consecutive threads -> consecutive addresses
    s += v; q += v * v;
  }
  part[(size_t)(blk * 2 + 0) * Mo + t] = s;
  part[(size_t)(blk * 2 + 1) * Mo + t] = q;
}

// ================================================================ 5b) BN1 finalize (1 block, deterministic order)
__global__ __launch_bounds__(256) void bn1_final(const float* __restrict__ part,
                                                 const float* __restrict__ g,
                                                 const float* __restrict__ bt,
                                                 float* __restrict__ scale,
                                                 float* __restrict__ shift) {
  int c = threadIdx.x;
  float s = 0.f, q = 0.f;
  for (int i = 0; i < NCHUNK; ++i) {
    s += part[(size_t)(i * 2 + 0) * Mo + c];
    q += part[(size_t)(i * 2 + 1) * Mo + c];
  }
  float m = s / (float)N1;
  float var = q / (float)N1 - m * m;
  float sc = g[c] * rsqrtf(var + EPS);
  scale[c] = sc;
  shift[c] = bt[c] - m * sc;       // b_dc cancels exactly through BN
}

// ================================================================ 6) BN1 + ReLU + pack channel-last bf16 hT[n][c]
__global__ __launch_bounds__(256) void bnrelu_pack(const float* __restrict__ out1,
                                                   const float* __restrict__ scale,
                                                   const float* __restrict__ shift,
                                                   unsigned short* __restrict__ hT) {
  int gid = blockIdx.x * blockDim.x + threadIdx.x;   // N1*Mo
  int c = gid & 255;
  float v = out1[gid];
  hT[gid] = f2bf(fmaxf(fmaf(scale[c], v, shift[c]), 0.f));
}

// ================================================================ 7) pack w_up per output parity: A2[p][o][t*256+c]
__global__ __launch_bounds__(256) void pack_wup(const float* __restrict__ w_up,
                                                unsigned short* __restrict__ A2) {
  int gid = blockIdx.x * blockDim.x + threadIdx.x;   // 4*Mo*K2
  if (gid >= 4 * Mo * K2) return;
  int p = gid / (Mo * K2);
  int r = gid % (Mo * K2);
  int o = r / K2, k = r % K2;
  int t = k >> 8, c = k & 255;
  int sy = p >> 1, sx = p & 1;
  int ty = t >> 1, tx = t & 1;
  int ky = 2 * ty + 1 - sy;        // parity-even output uses ky {1,3}; odd {0,2}
  int kx = 2 * tx + 1 - sx;
  A2[gid] = f2bf(w_up[(((size_t)c * Mo + o) * 4 + ky) * 4 + kx]);
}

// ================================================================ 8) GEMM2: transposed conv as 4 parity GEMMs, 32x32 tile per wave
__global__ __launch_bounds__(256) void gemm2_wmma(
    const unsigned short* __restrict__ A2,  // 4 x (Mo x K2) bf16
    const unsigned short* __restrict__ hT,  // N1 x Cc bf16, channel-last
    float* __restrict__ out) {              // (B, Mo, 96, 96) f32 (pre-BN2)
  const int wave = blockIdx.x * (blockDim.x >> 5) + (threadIdx.x >> 5);
  const int lane = threadIdx.x & 31;
  const int parity = wave / (MB1 * NB1);    // 4608 waves per parity
  int tw = wave % (MB1 * NB1);
  const int mt = tw % MB1;
  const int nt = tw / MB1;
  const int sy = parity >> 1, sx = parity & 1;
  const int l16 = lane & 15;
  const int hlf = lane >> 4;
  const int n0 = nt * 32 + l16, n1 = n0 + 16;
  const int b0 = n0 / HW, r0 = n0 % HW, jy0 = r0 / Ww, jx0 = r0 % Ww;
  const int b1 = n1 / HW, r1 = n1 % HW, jy1 = r1 / Ww, jx1 = r1 % Ww;
  const unsigned short* a0p = A2 + (size_t)parity * Mo * K2 +
                              (size_t)(mt * 32 + l16) * K2 + hlf * 16;
  const unsigned short* a1p = a0p + (size_t)16 * K2;
  v8f acc00 = {}, acc10 = {}, acc01 = {}, acc11 = {};
  for (int kb = 0; kb < K2; kb += 32) {
    int k16 = kb + hlf * 16;
    int t = k16 >> 8;                 // tap (16-chunk never crosses a tap)
    int c0 = k16 & 255;
    int ty = t >> 1, tx = t & 1;
    BFrag a0, a1, bf0, bf1;
    a0.u = *(const v16us*)(a0p + kb);
    a1.u = *(const v16us*)(a1p + kb);
    {
      int iy = jy0 + sy - ty, ix = jx0 + sx - tx;
      if (iy >= 0 && iy < Hh && ix >= 0 && ix < Ww) {
        const unsigned short* bp = hT + (size_t)(b0 * HW + iy * Ww + ix) * Cc + c0;
        bf0.u = *(const v16us*)bp;
        __builtin_prefetch(bp + 64, 0, 1);
      } else {
        bf0.u = (v16us)((unsigned short)0);
      }
    }
    {
      int iy = jy1 + sy - ty, ix = jx1 + sx - tx;
      if (iy >= 0 && iy < Hh && ix >= 0 && ix < Ww) {
        const unsigned short* bp = hT + (size_t)(b1 * HW + iy * Ww + ix) * Cc + c0;
        bf1.u = *(const v16us*)bp;
        __builtin_prefetch(bp + 64, 0, 1);
      } else {
        bf1.u = (v16us)((unsigned short)0);
      }
    }
    acc00 = WMMA_BF16(a0.b, bf0.b, acc00);
    acc10 = WMMA_BF16(a1.b, bf0.b, acc10);
    acc01 = WMMA_BF16(a0.b, bf1.b, acc01);
    acc11 = WMMA_BF16(a1.b, bf1.b, acc11);
  }
  const int mb = mt * 32 + hlf * 8;
  const int oy0 = 2 * jy0 + sy, ox0 = 2 * jx0 + sx;
  const int oy1 = 2 * jy1 + sy, ox1 = 2 * jx1 + sx;
#pragma unroll
  for (int r = 0; r < 8; ++r) {
    out[(((size_t)b0 * Mo + mb + r)      * HO + oy0) * WO + ox0] = acc00[r];
    out[(((size_t)b0 * Mo + mb + 16 + r) * HO + oy0) * WO + ox0] = acc10[r];
    out[(((size_t)b1 * Mo + mb + r)      * HO + oy1) * WO + ox1] = acc01[r];
    out[(((size_t)b1 * Mo + mb + 16 + r) * HO + oy1) * WO + ox1] = acc11[r];
  }
}

// ================================================================ 9) BN2 stats per channel (channel-major layout -> coalesced; LDS reduce)
__global__ __launch_bounds__(256) void bn2_stats(const float* __restrict__ out,
                                                 const float* __restrict__ g,
                                                 const float* __restrict__ bt,
                                                 float* __restrict__ scale,
                                                 float* __restrict__ shift) {
  __shared__ float ss[256], sq[256];
  int o = blockIdx.x, tid = threadIdx.x;
  float s = 0.f, q = 0.f;
  for (int b = 0; b < Bn; ++b) {
    const float* p = out + ((size_t)b * Mo + o) * (HO * WO);
    for (int i = tid; i < HO * WO; i += 256) { float v = p[i]; s += v; q += v * v; }
  }
  ss[tid] = s; sq[tid] = q;
  __syncthreads();
  for (int st = 128; st > 0; st >>= 1) {
    if (tid < st) { ss[tid] += ss[tid + st]; sq[tid] += sq[tid + st]; }
    __syncthreads();
  }
  if (tid == 0) {
    const float cnt = (float)(Bn * HO * WO);
    float m = ss[0] / cnt;
    float var = sq[0] / cnt - m * m;
    float sc = g[o] * rsqrtf(var + EPS);
    scale[o] = sc;
    shift[o] = bt[o] - m * sc;
  }
}

// ================================================================ 10) BN2 + ReLU in place
__global__ __launch_bounds__(256) void bnrelu2(float* __restrict__ out,
                                               const float* __restrict__ scale,
                                               const float* __restrict__ shift) {
  int gid = blockIdx.x * blockDim.x + threadIdx.x;   // Bn*Mo*HO*WO
  int c = (gid / (HO * WO)) & 255;
  float v = out[gid];
  out[gid] = fmaxf(fmaf(scale[c], v, shift[c]), 0.f);
}

// ================================================================ launcher
extern "C" void kernel_launch(void* const* d_in, const int* in_sizes, int n_in,
                              void* d_out, int out_size, void* d_ws, size_t ws_size,
                              hipStream_t stream) {
  const float* x     = (const float*)d_in[0];
  const float* w_off = (const float*)d_in[1];
  const float* b_off = (const float*)d_in[2];
  const float* w_mod = (const float*)d_in[3];
  const float* b_mod = (const float*)d_in[4];
  const float* w_dc  = (const float*)d_in[5];
  /* d_in[6] = b_dc: per-channel constant shift, cancels exactly through BN1 */
  const float* g1    = (const float*)d_in[7];
  const float* bt1   = (const float*)d_in[8];
  const float* w_up  = (const float*)d_in[9];
  const float* g2    = (const float*)d_in[10];
  const float* bt2   = (const float*)d_in[11];
  float* out = (float*)d_out;

  // workspace carving (256B aligned regions)
  char* base = (char*)d_ws;
  size_t off = 0;
  auto carve = [&](size_t bytes) {
    void* p = base + off;
    off += (bytes + 255) & ~(size_t)255;
    return p;
  };
  float*          offmod = (float*)carve((size_t)N1 * 27 * 4);          // ~2 MB
  unsigned short* V1     = (unsigned short*)carve((size_t)N1 * K1 * 2); // ~85 MB
  unsigned short* A1     = (unsigned short*)carve((size_t)Mo * K1 * 2); // ~1.2 MB
  float*          out1   = (float*)carve((size_t)N1 * Mo * 4);          // ~19 MB
  unsigned short* hT     = (unsigned short*)carve((size_t)N1 * Cc * 2); // ~9.4 MB
  unsigned short* A2     = (unsigned short*)carve((size_t)4 * Mo * K2 * 2);
  float*          part1  = (float*)carve((size_t)NCHUNK * 2 * Mo * 4);  // 147 KB
  float*          scale1 = (float*)carve(256 * 4);
  float*          shift1 = (float*)carve(256 * 4);
  float*          scale2 = (float*)carve(256 * 4);
  float*          shift2 = (float*)carve(256 * 4);
  if (ws_size < off) return;   // insufficient scratch: do nothing (deterministic)

  offmod_conv<<<(N1 * 27) / 256, 256, 0, stream>>>(x, w_off, b_off, w_mod, b_mod, offmod);
  build_v1<<<(N1 * 9) / 256, 256, 0, stream>>>(x, offmod, V1);
  pack_wdc<<<(Mo * K1) / 256, 256, 0, stream>>>(w_dc, A1);
  gemm1_wmma<<<(MB1 * NB1) / 8, 256, 0, stream>>>(A1, V1, out1);          // 576 blocks
  bn1_partial<<<NCHUNK, 256, 0, stream>>>(out1, part1);
  bn1_final<<<1, 256, 0, stream>>>(part1, g1, bt1, scale1, shift1);
  bnrelu_pack<<<(N1 * Mo) / 256, 256, 0, stream>>>(out1, scale1, shift1, hT);
  pack_wup<<<(4 * Mo * K2) / 256, 256, 0, stream>>>(w_up, A2);
  gemm2_wmma<<<(4 * MB1 * NB1) / 8, 256, 0, stream>>>(A2, hT, out);       // 2304 blocks
  bn2_stats<<<256, 256, 0, stream>>>(out, g2, bt2, scale2, shift2);
  bnrelu2<<<(Bn * Mo * HO * WO) / 256, 256, 0, stream>>>(out, scale2, shift2);
}
